// LongcatFlashMoE_43954695308103
// MI455X (gfx1250) — matrix operand order repaired and checked by reading, not compiled, not run on hardware
//
#include <hip/hip_runtime.h>
#include <hip/hip_bf16.h>

// ---------------- problem constants (fixed by reference) ----------------
constexpr int Bb   = 2;
constexpr int Ss   = 2048;
constexpr int Hh   = 1024;           // hidden
constexpr int Ii   = 512;            // expert ffn hidden
constexpr int E_R  = 32;             // routed experts
constexpr int E_T  = 40;             // total experts
constexpr int KK   = 6;              // topk
constexpr int Tt   = Bb * Ss;        // 4096 tokens
constexpr float SCALE = 2.5f;

constexpr int TM   = 256;            // token tile per workgroup (8 waves x 32 rows)
constexpr int AS   = Ii + 8;         // act LDS row stride in elems (16B pad -> conflict-free)
constexpr int ACT_BYTES = TM * AS * 2;            // 266,240
constexpr int BST_ROW   = 64 * 2 + 16;            // staged weight row: 64 K bf16 + 16B pad = 144
constexpr int BST_BYTES = 128 * BST_ROW;          // 18,432 per buffer (128 rows: 64 gate + 64 up)
constexpr int LDS_TOTAL = ACT_BYTES + 2 * BST_BYTES;   // 303,104 <= 320K

typedef unsigned short u16;
typedef unsigned int   u32;
typedef unsigned long long u64;

typedef __attribute__((ext_vector_type(16))) __bf16 v16bf;
typedef __attribute__((ext_vector_type(8)))  float  v8f;

union Frag {                          // one WMMA 16-bit operand fragment (8 VGPRs)
    uint4 q[2];
    v16bf v;
};

__device__ __forceinline__ u16 f2bf(float f) {
    u32 u = __float_as_uint(f);
    u32 r = (u + 0x7FFFu + ((u >> 16) & 1u)) >> 16;   // round-to-nearest-even
    return (u16)r;
}

// ---- CDNA5 async global->LDS copy (ASYNCcnt-tracked) ----
__device__ __forceinline__ void async_ld_b128(u32 ldsOff, u64 gaddr) {
    asm volatile("global_load_async_to_lds_b128 %0, %1, off"
                 :: "v"(ldsOff), "v"(gaddr) : "memory");
}
__device__ __forceinline__ void wait_async_le4() { asm volatile("s_wait_asynccnt 4" ::: "memory"); }
__device__ __forceinline__ void wait_async_0()   { asm volatile("s_wait_asynccnt 0" ::: "memory"); }

__device__ __forceinline__ v8f wmma_bf16(const v16bf& a, const v16bf& b, const v8f& c) {
    return __builtin_amdgcn_wmma_f32_16x16x32_bf16(false, a, false, b, (short)0, c, false, false);
}

// ---------------- fp32 -> bf16 bulk convert (vectorized) ----------------
__global__ __launch_bounds__(256) void cvt_bf16_kernel(const float* __restrict__ s,
                                                       u32* __restrict__ d, long n4) {
    long i = (long)blockIdx.x * blockDim.x + threadIdx.x;
    if (i >= n4) return;
    float4 f = reinterpret_cast<const float4*>(s)[i];
    u32 lo = (u32)f2bf(f.x) | ((u32)f2bf(f.y) << 16);
    u32 hi = (u32)f2bf(f.z) | ((u32)f2bf(f.w) << 16);
    reinterpret_cast<uint2*>(d)[i] = make_uint2(lo, hi);
}

// ---------------- router: logits -> softmax -> top-6 -> weights ----------------
__global__ __launch_bounds__(256) void router_kernel(const float* __restrict__ x,
                                                     const float* __restrict__ Wc,
                                                     const float* __restrict__ bias,
                                                     float* __restrict__ w_route,   // [E_R][T]
                                                     float* __restrict__ zero_w) {  // [T]
    const int lane = threadIdx.x & 31;
    const int wv   = threadIdx.x >> 5;
    const int t    = blockIdx.x * 8 + wv;           // one wave per token

    __shared__ float lg[8][48];
    __shared__ float tk[8][48];

    const float* xp = x + (size_t)t * Hh;
    float xr[32];
#pragma unroll
    for (int j = 0; j < 32; ++j) xr[j] = xp[j * 32 + lane];

    for (int e = 0; e < E_T; ++e) {
        const float* wp = Wc + (size_t)e * Hh;
        float acc = 0.f;
#pragma unroll
        for (int j = 0; j < 32; ++j) acc += xr[j] * wp[j * 32 + lane];
#pragma unroll
        for (int off = 16; off > 0; off >>= 1) acc += __shfl_down(acc, off, 32);
        if (lane == 0) lg[wv][e] = acc;
    }

    if (lane == 0) {
        float m = -3.4e38f;
        for (int e = 0; e < E_T; ++e) m = fmaxf(m, lg[wv][e]);
        float s = 0.f;
        for (int e = 0; e < E_T; ++e) { float p = __expf(lg[wv][e] - m); lg[wv][e] = p; s += p; }
        float inv = 1.f / s;
        for (int e = 0; e < E_T; ++e) tk[wv][e] = lg[wv][e] * inv + bias[e];
        for (int e = 0; e < E_R; ++e) w_route[(size_t)e * Tt + t] = 0.f;
        float zw = 0.f;
        for (int kk = 0; kk < KK; ++kk) {
            int best = 0; float bk = -3.4e38f;
            for (int e = 0; e < E_T; ++e) { float kv = tk[wv][e]; if (kv > bk) { bk = kv; best = e; } }
            tk[wv][best] = -3.4e38f;
            float w = lg[wv][best] * inv * SCALE;
            if (best < E_R) w_route[(size_t)best * Tt + t] = w;
            else            zw += w;
        }
        zero_w[t] = zw;
    }
}

// ---------------- zero-expert pass: out = x * zero_w (initializes d_out) ----------------
__global__ __launch_bounds__(256) void out0_kernel(const float* __restrict__ x,
                                                   const float* __restrict__ zero_w,
                                                   float* __restrict__ out) {
    size_t i = (size_t)blockIdx.x * blockDim.x + threadIdx.x;   // float4 index
    float4 v = reinterpret_cast<const float4*>(x)[i];
    int t = (int)((i * 4) >> 10);                               // /H
    float zw = zero_w[t];
    reinterpret_cast<float4*>(out)[i] =
        make_float4(v.x * zw, v.y * zw, v.z * zw, v.w * zw);
}

// ---------------- fused per-(token-tile, expert) SwiGLU MLP on WMMA ----------------
__global__ __launch_bounds__(256, 1)
__attribute__((amdgpu_waves_per_eu(2)))          // LDS caps us at 2 waves/SIMD anyway:
void expert_kernel(const u16* __restrict__ xbf,  // let the scheduler spend VGPRs on ILP
                   const u16* __restrict__ gup,   // [32][2I][H]
                   const u16* __restrict__ dwn,   // [32][H][I]
                   const float* __restrict__ w_route, // [E_R][T]
                   float* __restrict__ out) {     // [T][H]
    extern __shared__ char smem[];
    u16* act = reinterpret_cast<u16*>(smem);        // [TM][AS] bf16 activation tile
    const u32 smemBase = (u32)(uintptr_t)smem;      // LDS byte offset of dyn-shared base

    const int e    = blockIdx.y;
    const int t0   = blockIdx.x * TM;
    const int lane = threadIdx.x & 31;
    const int wv   = threadIdx.x >> 5;              // 8 waves: rows 32*wv .. 32*wv+31
    const int hi   = lane >> 4;                     // half-wave select
    const int ln   = lane & 15;
    const int kselA = hi * 8;                       // A-frag K sub-chunk base

    const u16* Ge = gup + (size_t)e * (2 * Ii) * Hh;
    const u16* De = dwn + (size_t)e * Hh * Ii;
    const u16* Arow0 = xbf + (size_t)(t0 + wv * 32 + ln) * Hh;   // row tile 0
    const u16* Arow1 = Arow0 + (size_t)16 * Hh;                  // row tile 1

    // ---- weight-stage assignment: 128 rows (64 gate + 64 up) x 64 K per stage ----
    const int sr = threadIdx.x >> 1;                // staged tile row 0..127
    const int sq = threadIdx.x & 1;                 // 64B half of the 128B row payload
    const u16* grow = Ge + (size_t)((sr < 64 ? 0 : Ii) + (sr & 63)) * Hh;
    const u32 ldsRowOff = smemBase + (u32)ACT_BYTES + sr * BST_ROW + sq * 64;

    // stage s: g = s>>4 (64-col group), k0 = (s&15)*64
    auto stage = [&](int s) {
        const int g  = s >> 4;
        const int k0 = (s & 15) * 64;
        const u64 gp = (u64)(uintptr_t)(grow + (size_t)g * 64 * Hh + k0 + sq * 32);
        const u32 lp = ldsRowOff + (u32)((s & 1) * BST_BYTES);
#pragma unroll
        for (int i = 0; i < 4; ++i)
            async_ld_b128(lp + i * 16, gp + (u64)i * 16);
    };

    // ================= phase 1: act = silu(x@Wg^T) * (x@Wu^T) =================
    v8f acc[2][4][2];                               // [row tile][ns][gate/up]
#pragma unroll
    for (int rt = 0; rt < 2; ++rt)
#pragma unroll
        for (int ns = 0; ns < 4; ++ns)
#pragma unroll
            for (int gu = 0; gu < 2; ++gu) acc[rt][ns][gu] = (v8f)(0.0f);

    stage(0);
    for (int s = 0; s < 128; ++s) {
        if (s + 1 < 128) { stage(s + 1); wait_async_le4(); }
        else             { wait_async_0(); }
        __syncthreads();                            // all waves' stage-s data in LDS

        const char* bbuf  = smem + ACT_BYTES + (s & 1) * BST_BYTES;
        const int   kbase = (s & 15) * 64;

        // hoist all A fragments for this stage (one clause; in-order returns
        // overlap the early WMMAs)
        Frag aF[2][2];                              // [ksub][row tile]
#pragma unroll
        for (int ksub = 0; ksub < 2; ++ksub) {
            const u16* ap0 = Arow0 + kbase + ksub * 32 + kselA;
            const u16* ap1 = Arow1 + kbase + ksub * 32 + kselA;
            aF[ksub][0].q[0] = *reinterpret_cast<const uint4*>(ap0);
            aF[ksub][0].q[1] = *reinterpret_cast<const uint4*>(ap0 + 16);
            aF[ksub][1].q[0] = *reinterpret_cast<const uint4*>(ap1);
            aF[ksub][1].q[1] = *reinterpret_cast<const uint4*>(ap1 + 16);
        }

        // B-fragment pair pipeline: (gate,up) for pair p = (ksub = p>>2, ns = p&3)
        auto loadBpair = [&](int p, Frag& bg, Frag& bu) {
            const int ksub = p >> 2, ns = p & 3;
            const char* base = bbuf + (ns * 16 + ln) * BST_ROW + ksub * 64 + hi * 16;
            bg.q[0] = *reinterpret_cast<const uint4*>(base);
            bg.q[1] = *reinterpret_cast<const uint4*>(base + 32);
            const char* bupp = base + 64 * BST_ROW;
            bu.q[0] = *reinterpret_cast<const uint4*>(bupp);
            bu.q[1] = *reinterpret_cast<const uint4*>(bupp + 32);
        };

        // 3-slot ring, prefetch distance 2: while consuming pair p, pairs
        // p+1 and p+2 are in flight -> waits become dscnt<=8 instead of 0.
        Frag bg[3], bu[3];
        loadBpair(0, bg[0], bu[0]);
        loadBpair(1, bg[1], bu[1]);
#pragma unroll
        for (int p = 0; p < 8; ++p) {
            if (p + 2 < 8) loadBpair(p + 2, bg[(p + 2) % 3], bu[(p + 2) % 3]);
            const int ksub = p >> 2, ns = p & 3, sl = p % 3;
            acc[0][ns][0] = wmma_bf16(aF[ksub][0].v, bg[sl].v, acc[0][ns][0]);
            acc[1][ns][0] = wmma_bf16(aF[ksub][1].v, bg[sl].v, acc[1][ns][0]);
            acc[0][ns][1] = wmma_bf16(aF[ksub][0].v, bu[sl].v, acc[0][ns][1]);
            acc[1][ns][1] = wmma_bf16(aF[ksub][1].v, bu[sl].v, acc[1][ns][1]);
        }
        __syncthreads();                            // done consuming before buffer reuse

        if ((s & 15) == 15) {                       // 64-col group complete -> SwiGLU
            const int g = s >> 4;
#pragma unroll
            for (int rt = 0; rt < 2; ++rt)
#pragma unroll
                for (int ns = 0; ns < 4; ++ns) {
                    const int col = g * 64 + ns * 16 + ln;
#pragma unroll
                    for (int j = 0; j < 8; ++j) {
                        float gt = acc[rt][ns][0][j];
                        float up = acc[rt][ns][1][j];
                        float av = (gt / (1.f + __expf(-gt))) * up;
                        const int m = wv * 32 + rt * 16 + hi * 8 + j;
                        act[(size_t)m * AS + col] = f2bf(av);
                        acc[rt][ns][0][j] = 0.f;
                        acc[rt][ns][1][j] = 0.f;
                    }
                }
        }
    }
    __syncthreads();

    // ================= phase 2: y = act @ Wd^T ; out += wgt * y =================
    const int baseRow0 = t0 + wv * 32 + hi * 8;     // rows for c0 (j -> +j)
    const int baseRow1 = baseRow0 + 16;             // rows for c1
    float wgt0[8], wgt1[8];
#pragma unroll
    for (int j = 0; j < 8; ++j) {
        wgt0[j] = w_route[(size_t)e * Tt + baseRow0 + j];
        wgt1[j] = w_route[(size_t)e * Tt + baseRow1 + j];
    }

    const u16* actR0 = act + (size_t)(wv * 32 + ln) * AS;
    const u16* actR1 = actR0 + (size_t)16 * AS;
    for (int n0 = 0; n0 < Hh; n0 += 16) {
        v8f c0 = (v8f)(0.0f), c1 = (v8f)(0.0f);
        const int h = n0 + ln;
        const u16* Brow = De + (size_t)h * Ii;

        Frag A0[3], A1[3], Bf[3];
        auto ldP2 = [&](int kk, int slot) {
            const int k0 = kk * 32;
            const u16* bp = Brow + k0 + hi * 8;
            Bf[slot].q[0] = *reinterpret_cast<const uint4*>(bp);
            Bf[slot].q[1] = *reinterpret_cast<const uint4*>(bp + 16);
            const u16* ap0 = actR0 + k0 + kselA;
            const u16* ap1 = actR1 + k0 + kselA;
            A0[slot].q[0] = *reinterpret_cast<const uint4*>(ap0);
            A0[slot].q[1] = *reinterpret_cast<const uint4*>(ap0 + 16);
            A1[slot].q[0] = *reinterpret_cast<const uint4*>(ap1);
            A1[slot].q[1] = *reinterpret_cast<const uint4*>(ap1 + 16);
        };

        ldP2(0, 0);
        ldP2(1, 1);
#pragma unroll
        for (int kk = 0; kk < 16; ++kk) {
            if (kk + 2 < 16) ldP2(kk + 2, (kk + 2) % 3);   // distance-2 prefetch
            const int sl = kk % 3;
            c0 = wmma_bf16(A0[sl].v, Bf[sl].v, c0);
            c1 = wmma_bf16(A1[sl].v, Bf[sl].v, c1);
        }
#pragma unroll
        for (int j = 0; j < 8; ++j) {
            if (wgt0[j] != 0.f)
                atomicAdd(&out[(size_t)(baseRow0 + j) * Hh + h], wgt0[j] * c0[j]);
            if (wgt1[j] != 0.f)
                atomicAdd(&out[(size_t)(baseRow1 + j) * Hh + h], wgt1[j] * c1[j]);
        }
    }
}

// ---------------- host-side launch ----------------
extern "C" void kernel_launch(void* const* d_in, const int* in_sizes, int n_in,
                              void* d_out, int out_size, void* d_ws, size_t ws_size,
                              hipStream_t stream) {
    const float* x    = (const float*)d_in[0];   // [B,S,H]
    const float* Wc   = (const float*)d_in[1];   // [E_T,H]
    const float* bias = (const float*)d_in[2];   // [E_T]
    const float* gup  = (const float*)d_in[3];   // [E_T,2I,H]
    const float* dwn  = (const float*)d_in[4];   // [E_R,H,I]
    float* out = (float*)d_out;

    char* ws = (char*)d_ws;
    size_t off = 0;
    auto take = [&](size_t bytes) { char* p = ws + off; off = (off + bytes + 255) & ~(size_t)255; return p; };
    u16*   xbf     = (u16*)  take((size_t)Tt * Hh * 2);                 // 8 MB
    u16*   gupbf   = (u16*)  take((size_t)E_R * 2 * Ii * Hh * 2);       // 64 MB
    u16*   dwnbf   = (u16*)  take((size_t)E_R * Hh * Ii * 2);           // 32 MB
    float* w_route = (float*)take((size_t)E_R * Tt * 4);                // 512 KB
    float* zero_w  = (float*)take((size_t)Tt * 4);                      // 16 KB
    (void)ws_size; (void)in_sizes; (void)n_in; (void)out_size;

    {   long n4 = (long)Tt * Hh / 4;
        cvt_bf16_kernel<<<(n4 + 255) / 256, 256, 0, stream>>>(x, (u32*)xbf, n4); }
    {   long n4 = (long)E_R * 2 * Ii * Hh / 4;   // first 32 experts of gate_up
        cvt_bf16_kernel<<<(n4 + 255) / 256, 256, 0, stream>>>(gup, (u32*)gupbf, n4); }
    {   long n4 = (long)E_R * Hh * Ii / 4;
        cvt_bf16_kernel<<<(n4 + 255) / 256, 256, 0, stream>>>(dwn, (u32*)dwnbf, n4); }

    router_kernel<<<Tt / 8, 256, 0, stream>>>(x, Wc, bias, w_route, zero_w);
    out0_kernel<<<((size_t)Tt * Hh / 4) / 256, 256, 0, stream>>>(x, zero_w, out);

    dim3 grid(Tt / TM, E_R);
    expert_kernel<<<grid, 256, (size_t)LDS_TOTAL, stream>>>(xbf, gupbf, dwnbf, w_route, out);
}
